// AttentionDecoder_48206712930895
// MI455X (gfx1250) — matrix operand order repaired and checked by reading, not compiled, not run on hardware
//
#include <hip/hip_runtime.h>
#include <hip/hip_bf16.h>

// ---------------------------------------------------------------------------
// AttentionDecoder for MI455X (gfx1250, wave32, WMMA).
// Fixed problem: B=64, S=512, T=96, E=1024, NH=16, DH=64, O=1.
// Strategy:
//   * one-time: convert encoder + all weights fp32 -> bf16 (L2-resident),
//     project K/V with a bf16 WMMA GEMM into [B,NH,S,DH] bf16 caches
//     (halves the dominant HBM traffic: 96 steps x (K+V) reads).
//   * per step (sequential dependency, 5 kernels/step on the stream):
//       q = (h @ wq^T + bq) * 1/sqrt(DH)          (WMMA bf16 -> f32)
//       per-(b,head) softmax-attention            (VALU + LDS reductions)
//       attn_out = ctx @ out_proj^T + b           (WMMA bf16 -> bf16)
//       gates = attn_out @ w_ih^T + h @ w_hh^T +b (fused dual-A WMMA -> f32)
//       LSTM cell + h2 . w_out dot                (VALU + trans + LDS reduce)
// ---------------------------------------------------------------------------

typedef __bf16 bf16_t;
typedef __attribute__((ext_vector_type(16))) __bf16 v16bf;
typedef __attribute__((ext_vector_type(8)))  __bf16 v8bf;
typedef __attribute__((ext_vector_type(8)))  float  v8f;

static constexpr int Bsz = 64;
static constexpr int Ssz = 512;
static constexpr int Tsz = 96;
static constexpr int Esz = 1024;
static constexpr int NH  = 16;
static constexpr int DH  = 64;

// ---------------- WMMA fragment loaders (16x16x32 bf16) --------------------
// A (16x32, MxK): lanes 0-15 hold row M=lane, K chunks {0..7,16..23};
//                 lanes 16-31 hold row M=lane-16, K chunks {8..15,24..31}.
__device__ __forceinline__ v16bf load_a_frag(const bf16_t* __restrict__ A,
                                             int lda, int m0, int k0,
                                             int lid, int half) {
  const bf16_t* base = A + (size_t)(m0 + lid) * lda + k0 + half * 8;
  v8bf lo = *reinterpret_cast<const v8bf*>(base);
  v8bf hi = *reinterpret_cast<const v8bf*>(base + 16);
  return __builtin_shufflevector(lo, hi, 0,1,2,3,4,5,6,7,8,9,10,11,12,13,14,15);
}

// B (32x16, KxN) with B[k,n] = W[n,k], W row-major [N,K]:
// lanes 0-15: col N=lane, K {0..7,16..23}; lanes 16-31: col N=lane-16, K {8..15,24..31}.
__device__ __forceinline__ v16bf load_b_frag(const bf16_t* __restrict__ W,
                                             int ldw, int n0, int k0,
                                             int lid, int half) {
  const bf16_t* base = W + (size_t)(n0 + lid) * ldw + k0 + half * 8;
  v8bf lo = *reinterpret_cast<const v8bf*>(base);
  v8bf hi = *reinterpret_cast<const v8bf*>(base + 16);
  return __builtin_shufflevector(lo, hi, 0,1,2,3,4,5,6,7,8,9,10,11,12,13,14,15);
}

// ---------------- GEMM: C = (A1 W1^T [+ A2 W2^T] + bias) * alpha -----------
// W row-major [N,K]; per-wave tile 16 x (16*NT); fp32 accum via WMMA.
template <int NT>
__global__ __launch_bounds__(256) void gemm_bf16_wmma(
    const bf16_t* __restrict__ A1, const bf16_t* __restrict__ W1,
    const bf16_t* __restrict__ A2, const bf16_t* __restrict__ W2,
    const float* __restrict__ bias, float alpha,
    float* __restrict__ outF, bf16_t* __restrict__ outB,
    int M, int N, int Kd, int kv_remap) {
  const int wave = blockIdx.x * (blockDim.x >> 5) + (threadIdx.x >> 5);
  const int lane = threadIdx.x & 31;
  const int lid  = lane & 15;
  const int half = lane >> 4;
  const int ntiles = N / (16 * NT);
  const int mtiles = M / 16;
  if (wave >= mtiles * ntiles) return;
  const int m0 = (wave / ntiles) * 16;
  const int n0 = (wave % ntiles) * 16 * NT;

  v8f acc[NT] = {};

  for (int k0 = 0; k0 < Kd; k0 += 32) {
    if (k0 + 32 < Kd)  // gfx1250 global_prefetch_b8 for next K-slab of A
      __builtin_prefetch(A1 + (size_t)(m0 + lid) * Kd + k0 + 32 + half * 8, 0, 1);
    v16bf a = load_a_frag(A1, Kd, m0, k0, lid, half);
#pragma unroll
    for (int t = 0; t < NT; ++t) {
      v16bf b = load_b_frag(W1, Kd, n0 + t * 16, k0, lid, half);
      acc[t] = __builtin_amdgcn_wmma_f32_16x16x32_bf16(
          false, a, false, b, (short)0, acc[t], false, false);
    }
  }
  if (A2 != nullptr) {
    for (int k0 = 0; k0 < Kd; k0 += 32) {
      v16bf a = load_a_frag(A2, Kd, m0, k0, lid, half);
#pragma unroll
      for (int t = 0; t < NT; ++t) {
        v16bf b = load_b_frag(W2, Kd, n0 + t * 16, k0, lid, half);
        acc[t] = __builtin_amdgcn_wmma_f32_16x16x32_bf16(
            false, a, false, b, (short)0, acc[t], false, false);
      }
    }
  }

  // C layout: VGPR r -> (m0 + half*8 + r, n0 + t*16 + lid)
  const int mrow = m0 + half * 8;
#pragma unroll
  for (int t = 0; t < NT; ++t) {
    const int n = n0 + t * 16 + lid;
    const float bv = bias ? bias[n] : 0.0f;
#pragma unroll
    for (int r = 0; r < 8; ++r) {
      const int m = mrow + r;
      const float v = (acc[t][r] + bv) * alpha;
      if (!kv_remap) {
        const size_t idx = (size_t)m * N + n;
        if (outF) outF[idx] = v;
        if (outB) outB[idx] = (bf16_t)v;
      } else {  // row m=(b,s), col n=(h,d) -> [B,NH,S,DH] bf16 cache
        const int bb = m >> 9, s = m & 511, hh = n >> 6, d = n & 63;
        outB[(((size_t)(bb * NH + hh)) * Ssz + s) * DH + d] = (bf16_t)v;
      }
    }
  }
}

// ---------------- per-(b,head) attention -----------------------------------
__global__ __launch_bounds__(256) void attn_kernel(
    const float* __restrict__ q, const bf16_t* __restrict__ Kc,
    const bf16_t* __restrict__ Vc, bf16_t* __restrict__ ctx,
    float* __restrict__ attnw, int write_w) {
  __shared__ float qs[DH];
  __shared__ float sc[Ssz];
  __shared__ float red[256];
  __shared__ float part[4][DH];
  const int b = blockIdx.x >> 4;
  const int h = blockIdx.x & 15;
  const int tid = threadIdx.x;
  if (tid < DH) qs[tid] = q[(size_t)b * Esz + h * DH + tid];
  __syncthreads();

  const size_t head_base = ((size_t)(b * NH + h)) * Ssz * DH;
  const bf16_t* Kb = Kc + head_base;

  float lmax = -1e30f;
  for (int s = tid; s < Ssz; s += 256) {
    const bf16_t* kr = Kb + (size_t)s * DH;
    float dot = 0.0f;
#pragma unroll
    for (int c = 0; c < 8; ++c) {
      v8bf kv = *reinterpret_cast<const v8bf*>(kr + c * 8);
#pragma unroll
      for (int j = 0; j < 8; ++j) dot += qs[c * 8 + j] * (float)kv[j];
    }
    sc[s] = dot;
    lmax = fmaxf(lmax, dot);
  }
  red[tid] = lmax;
  __syncthreads();
  for (int w = 128; w > 0; w >>= 1) {
    if (tid < w) red[tid] = fmaxf(red[tid], red[tid + w]);
    __syncthreads();
  }
  const float mx = red[0];
  __syncthreads();

  float lsum = 0.0f;
  for (int s = tid; s < Ssz; s += 256) {
    const float e = __expf(sc[s] - mx);
    sc[s] = e;
    lsum += e;
  }
  red[tid] = lsum;
  __syncthreads();
  for (int w = 128; w > 0; w >>= 1) {
    if (tid < w) red[tid] += red[tid + w];
    __syncthreads();
  }
  const float inv = 1.0f / red[0];
  __syncthreads();

  // ctx[d] = (1/Z) * sum_s exp(s) * V[s,d]
  const bf16_t* Vb = Vc + head_base;
  const int d = tid & (DH - 1);
  const int g = tid >> 6;
  float accv = 0.0f;
  for (int s = g * 128; s < g * 128 + 128; ++s)
    accv += sc[s] * (float)Vb[(size_t)s * DH + d];
  part[g][d] = accv;
  __syncthreads();
  if (tid < DH) {
    const float cv = (part[0][tid] + part[1][tid] + part[2][tid] + part[3][tid]) * inv;
    ctx[(size_t)b * Esz + h * DH + tid] = (bf16_t)cv;
  }
  if (write_w) {  // last step: mean over heads -> attn_weights [B,1,S]
    for (int s = tid; s < Ssz; s += 256)
      atomicAdd(&attnw[(size_t)b * Ssz + s], sc[s] * inv * (1.0f / NH));
  }
}

// ---------------- LSTM cell + output projection ----------------------------
__global__ __launch_bounds__(256) void lstm_kernel(
    const float* __restrict__ gates, float* __restrict__ c,
    bf16_t* __restrict__ h_bf, const float* __restrict__ w_out,
    const float* __restrict__ b_out, float* __restrict__ out, int t) {
  __shared__ float red[256];
  const int b = blockIdx.x;
  const int tid = threadIdx.x;
  const float* gb = gates + (size_t)b * (4 * Esz);
  float dot = 0.0f;
#pragma unroll
  for (int r = 0; r < 4; ++r) {
    const int e = tid + r * 256;
    const float gi = gb[e];
    const float gf = gb[Esz + e];
    const float gg = gb[2 * Esz + e];
    const float go = gb[3 * Esz + e];
    const float si = 1.0f / (1.0f + __expf(-gi));
    const float sf = 1.0f / (1.0f + __expf(-gf));
    const float so = 1.0f / (1.0f + __expf(-go));
    const size_t idx = (size_t)b * Esz + e;
    const float c2 = sf * c[idx] + si * tanhf(gg);
    const float h2 = so * tanhf(c2);
    c[idx] = c2;
    h_bf[idx] = (bf16_t)h2;
    dot += h2 * w_out[e];
  }
  red[tid] = dot;
  __syncthreads();
  for (int w = 128; w > 0; w >>= 1) {
    if (tid < w) red[tid] += red[tid + w];
    __syncthreads();
  }
  if (tid == 0) out[(size_t)b * Tsz + t] = red[0] + b_out[0];
}

// ---------------- small helpers --------------------------------------------
__global__ void cvt_f32_bf16(const float* __restrict__ in,
                             bf16_t* __restrict__ out, size_t n) {
  const size_t i = (size_t)blockIdx.x * blockDim.x + threadIdx.x;
  if (i < n) out[i] = (bf16_t)in[i];
}

__global__ void bias_sum_kernel(const float* __restrict__ a,
                                const float* __restrict__ b,
                                float* __restrict__ o, int n) {
  const int i = blockIdx.x * blockDim.x + threadIdx.x;
  if (i < n) o[i] = a[i] + b[i];
}

// ---------------- host-side launchers --------------------------------------
template <int NT>
static void gemm_launch(hipStream_t stream, const bf16_t* A1, const bf16_t* W1,
                        const bf16_t* A2, const bf16_t* W2, const float* bias,
                        float alpha, float* outF, bf16_t* outB, int M, int N,
                        int Kd, int remap) {
  const int waves = (M / 16) * (N / (16 * NT));
  const int blocks = (waves + 7) / 8;  // 8 wave32 per 256-thread block
  gemm_bf16_wmma<NT><<<blocks, 256, 0, stream>>>(A1, W1, A2, W2, bias, alpha,
                                                 outF, outB, M, N, Kd, remap);
}

extern "C" void kernel_launch(void* const* d_in, const int* in_sizes, int n_in,
                              void* d_out, int out_size, void* d_ws,
                              size_t ws_size, hipStream_t stream) {
  const float* enc        = (const float*)d_in[0];   // [64,512,1024]
  // d_in[1] = decoder_inputs: unused by the reference computation.
  const float* in_proj_w  = (const float*)d_in[2];   // [3E,E]
  const float* in_proj_b  = (const float*)d_in[3];   // [3E]
  const float* out_proj_w = (const float*)d_in[4];   // [E,E]
  const float* out_proj_b = (const float*)d_in[5];   // [E]
  const float* w_ih       = (const float*)d_in[6];   // [4E,E]
  const float* w_hh       = (const float*)d_in[7];   // [4E,E]
  const float* b_ih       = (const float*)d_in[8];   // [4E]
  const float* b_hh       = (const float*)d_in[9];   // [4E]
  const float* w_out      = (const float*)d_in[10];  // [1,E]
  const float* b_out      = (const float*)d_in[11];  // [1]

  float* out_main = (float*)d_out;            // outputs [B,T,1] = 64*96
  float* out_attn = out_main + Bsz * Tsz;     // attn_weights [B,1,S] = 64*512

  // ---- workspace carve-up -------------------------------------------------
  char* ws = (char*)d_ws;
  size_t off = 0;
  auto take = [&](size_t bytes) -> void* {
    void* p = ws + off;
    off += (bytes + 255) & ~(size_t)255;
    return p;
  };
  const size_t nEnc = (size_t)Bsz * Ssz * Esz;             // 33.5M
  bf16_t* enc_bf    = (bf16_t*)take(nEnc * 2);
  bf16_t* inproj_bf = (bf16_t*)take((size_t)3 * Esz * Esz * 2);
  bf16_t* outp_bf   = (bf16_t*)take((size_t)Esz * Esz * 2);
  bf16_t* wih_bf    = (bf16_t*)take((size_t)4 * Esz * Esz * 2);
  bf16_t* whh_bf    = (bf16_t*)take((size_t)4 * Esz * Esz * 2);
  bf16_t* Kc        = (bf16_t*)take(nEnc * 2);             // [B,NH,S,DH]
  bf16_t* Vc        = (bf16_t*)take(nEnc * 2);
  float*  q_f       = (float*)take((size_t)Bsz * Esz * 4);
  bf16_t* ctx_bf    = (bf16_t*)take((size_t)Bsz * Esz * 2);
  bf16_t* attn_bf   = (bf16_t*)take((size_t)Bsz * Esz * 2);
  float*  gates_f   = (float*)take((size_t)Bsz * 4 * Esz * 4);
  float*  c_f       = (float*)take((size_t)Bsz * Esz * 4);
  bf16_t* h_bf      = (bf16_t*)take((size_t)Bsz * Esz * 2);
  float*  bias_sum  = (float*)take((size_t)4 * Esz * 4);
  (void)ws_size; (void)in_sizes; (void)n_in; (void)out_size;

  // ---- one-time setup -----------------------------------------------------
  auto cvt = [&](const float* src, bf16_t* dst, size_t n) {
    cvt_f32_bf16<<<(unsigned)((n + 255) / 256), 256, 0, stream>>>(src, dst, n);
  };
  cvt(enc, enc_bf, nEnc);
  cvt(in_proj_w, inproj_bf, (size_t)3 * Esz * Esz);
  cvt(out_proj_w, outp_bf, (size_t)Esz * Esz);
  cvt(w_ih, wih_bf, (size_t)4 * Esz * Esz);
  cvt(w_hh, whh_bf, (size_t)4 * Esz * Esz);
  bias_sum_kernel<<<16, 256, 0, stream>>>(b_ih, b_hh, bias_sum, 4 * Esz);

  hipMemsetAsync(h_bf, 0, (size_t)Bsz * Esz * 2, stream);     // h0 = 0
  hipMemsetAsync(c_f, 0, (size_t)Bsz * Esz * 4, stream);      // c0 = 0
  hipMemsetAsync(out_attn, 0, (size_t)Bsz * Ssz * 4, stream); // atomics target

  // K/V projection: [B*S, E] @ wk/wv^T -> bf16 caches [B,NH,S,DH]
  gemm_launch<4>(stream, enc_bf, inproj_bf + (size_t)1 * Esz * Esz, nullptr,
                 nullptr, in_proj_b + Esz, 1.0f, nullptr, Kc, Bsz * Ssz, Esz,
                 Esz, 1);
  gemm_launch<4>(stream, enc_bf, inproj_bf + (size_t)2 * Esz * Esz, nullptr,
                 nullptr, in_proj_b + 2 * Esz, 1.0f, nullptr, Vc, Bsz * Ssz,
                 Esz, Esz, 1);

  const float scale = 0.125f;  // 1/sqrt(DH=64)

  // ---- sequential decode loop --------------------------------------------
  for (int t = 0; t < Tsz; ++t) {
    // q = (h @ wq^T + bq) * scale
    gemm_launch<2>(stream, h_bf, inproj_bf, nullptr, nullptr, in_proj_b, scale,
                   q_f, nullptr, Bsz, Esz, Esz, 0);
    // softmax attention per (b, head)
    attn_kernel<<<Bsz * NH, 256, 0, stream>>>(q_f, Kc, Vc, ctx_bf, out_attn,
                                              t == Tsz - 1 ? 1 : 0);
    // attn_out = ctx @ out_proj^T + b
    gemm_launch<2>(stream, ctx_bf, outp_bf, nullptr, nullptr, out_proj_b, 1.0f,
                   nullptr, attn_bf, Bsz, Esz, Esz, 0);
    // gates = attn_out @ w_ih^T + h @ w_hh^T + (b_ih + b_hh)
    gemm_launch<2>(stream, attn_bf, wih_bf, h_bf, whh_bf, bias_sum, 1.0f,
                   gates_f, nullptr, Bsz, 4 * Esz, Esz, 0);
    // LSTM cell + out = h2 . w_out + b_out
    lstm_kernel<<<Bsz, 256, 0, stream>>>(gates_f, c_f, h_bf, w_out, b_out,
                                         out_main, t);
  }
}
